// Attention_83743272337758
// MI455X (gfx1250) — compile-verified, matrix operand order checked by reading, not run on hardware
//
#include <hip/hip_runtime.h>
#include <hip/hip_bf16.h>

// ---------- types ----------
typedef __attribute__((ext_vector_type(16))) __bf16 v16bf;
typedef __attribute__((ext_vector_type(8)))  float  v8f;
typedef int v4i __attribute__((vector_size(16)));  // matches builtin param type

// ---------- bf16 helpers (RNE) ----------
__device__ __forceinline__ unsigned short f2bf(float f) {
  unsigned u = __builtin_bit_cast(unsigned, f);
  u += 0x7FFFu + ((u >> 16) & 1u);
  return (unsigned short)(u >> 16);
}
__device__ __forceinline__ unsigned pack2bf(float a, float b) {
  return (unsigned)f2bf(a) | ((unsigned)f2bf(b) << 16);
}
__device__ __forceinline__ float bf2f(unsigned short h) {
  unsigned u = (unsigned)h << 16;
  return __builtin_bit_cast(float, u);
}

// ---------- async global->LDS copy (gfx1250), with safe fallback ----------
#if __has_builtin(__builtin_amdgcn_global_load_async_to_lds_b128) && \
    __has_builtin(__builtin_amdgcn_s_wait_asynccnt)
#define HAVE_ASYNC_LDS 1
#else
#define HAVE_ASYNC_LDS 0
#endif

__device__ __forceinline__ void async_cp16(const void* g, void* l) {
#if HAVE_ASYNC_LDS
  __builtin_amdgcn_global_load_async_to_lds_b128(
      (__attribute__((address_space(1))) v4i*)g,
      (__attribute__((address_space(3))) v4i*)l, 0, 0);
#else
  *(uint4*)l = *(const uint4*)g;
#endif
}
__device__ __forceinline__ void async_wait() {
#if HAVE_ASYNC_LDS
  __builtin_amdgcn_s_wait_asynccnt(0);
#endif
}

// ---------- LDS fragment load (row stride 72 bf16 = 144B, 16B aligned) ----------
union FragU { uint4 u[2]; v16bf v; };
__device__ __forceinline__ v16bf ld_frag72(const unsigned short* s, int row, int hi) {
  FragU f;
  f.u[0] = *(const uint4*)(s + row * 72 + hi * 8);       // K =  hi*8 ..  +7
  f.u[1] = *(const uint4*)(s + row * 72 + 16 + hi * 8);  // K = 16+hi*8 .. +7
  return f.v;
}

// ---------------------------------------------------------------------------
// WMMA GEMM: C = A * B (*alpha on store).
//  A_F32/B_F32: operand dtype in memory (fp32 converted to bf16 while staging)
//  BT         : B stored [N][K] row-major (q*k^T) vs [K][N]
//  OUT_BF16   : bf16 store (workspace) vs fp32 (final output)
// Block: 256 threads = 8 waves (4x2); tile 256x128; wave tile 64x64 = 16 WMMA;
// BK=64 (2 K-blocks of 32 per stage). M%256==0, N%128==0, K%64==0.
// ---------------------------------------------------------------------------
template <int A_F32, int B_F32, int BT, int OUT_BF16>
__global__ __launch_bounds__(256) void gemm_wmma(
    const void* __restrict__ Ap, long long sAb, int lda,
    const void* __restrict__ Bp, long long sBb, int ldb,
    void* __restrict__ Cp, long long sCb, int ldc,
    int K, float alpha) {
  static_assert(!(BT && B_F32), "BT path implemented for bf16 B only");
  __shared__ alignas(16) unsigned short As[256 * 72];  // 36 KiB
  __shared__ alignas(16) unsigned short Bs[128 * 72];  // 18 KiB

  const int tid = threadIdx.x;
  const int bx = blockIdx.x, by = blockIdx.y, bz = blockIdx.z;

  const float* Af = nullptr;
  const unsigned short* Ah = nullptr;
  if constexpr (A_F32)
    Af = (const float*)Ap + (long long)bz * sAb + (long long)by * 256 * lda;
  else
    Ah = (const unsigned short*)Ap + (long long)bz * sAb + (long long)by * 256 * lda;

  const long long boff =
      (long long)bz * sBb + (BT ? (long long)bx * 128 * ldb : (long long)(bx * 128));
  const float* Bf = nullptr;
  const unsigned short* Bh = nullptr;
  if constexpr (B_F32)
    Bf = (const float*)Bp + boff;
  else
    Bh = (const unsigned short*)Bp + boff;

  const int wave = tid >> 5, lane = tid & 31;
  const int wm = wave >> 1, wn = wave & 1;  // 4x2 wave grid
  const int l15 = lane & 15, hi = lane >> 4;

  v8f acc[4][4];
#pragma unroll
  for (int i = 0; i < 4; ++i)
#pragma unroll
    for (int j = 0; j < 4; ++j) acc[i][j] = 0.0f;

  for (int k0 = 0; k0 < K; k0 += 64) {
    // ---- stage A tile: 256 rows x 64 K ----
    if constexpr (A_F32) {
#pragma unroll
      for (int it = 0; it < 16; ++it) {
        int idx = (it * 256 + tid) * 4;
        int r = idx >> 6, kk = idx & 63;
        float4 v = *(const float4*)(Af + (long long)r * lda + k0 + kk);
        uint2 p;
        p.x = pack2bf(v.x, v.y);
        p.y = pack2bf(v.z, v.w);
        *(uint2*)&As[r * 72 + kk] = p;
      }
    } else {
#pragma unroll
      for (int it = 0; it < 8; ++it) {
        int idx = it * 256 + tid;
        int r = idx >> 3, seg = (idx & 7) * 8;
        async_cp16(Ah + (long long)r * lda + k0 + seg, &As[r * 72 + seg]);
      }
    }
    // ---- stage B tile ----
    if constexpr (BT) {
      // [N][K] row-major bf16: straight async copy, 128 rows x 64
#pragma unroll
      for (int it = 0; it < 4; ++it) {
        int idx = it * 256 + tid;
        int r = idx >> 3, seg = (idx & 7) * 8;
        async_cp16(Bh + (long long)r * ldb + k0 + seg, &Bs[r * 72 + seg]);
      }
    } else {
      // [K][N] row-major: transpose into n-major LDS, 64 k-rows x 128 n
#pragma unroll
      for (int it = 0; it < 8; ++it) {
        int idx = (it * 256 + tid) * 4;
        int kk = idx >> 7, n = idx & 127;
        if constexpr (B_F32) {
          float4 v = *(const float4*)(Bf + (long long)(k0 + kk) * ldb + n);
          Bs[(n + 0) * 72 + kk] = f2bf(v.x);
          Bs[(n + 1) * 72 + kk] = f2bf(v.y);
          Bs[(n + 2) * 72 + kk] = f2bf(v.z);
          Bs[(n + 3) * 72 + kk] = f2bf(v.w);
        } else {
          uint2 v = *(const uint2*)(Bh + (long long)(k0 + kk) * ldb + n);
          Bs[(n + 0) * 72 + kk] = (unsigned short)(v.x & 0xffffu);
          Bs[(n + 1) * 72 + kk] = (unsigned short)(v.x >> 16);
          Bs[(n + 2) * 72 + kk] = (unsigned short)(v.y & 0xffffu);
          Bs[(n + 3) * 72 + kk] = (unsigned short)(v.y >> 16);
        }
      }
    }
    async_wait();
    __syncthreads();

    // ---- 2 K-blocks of 32, 16 WMMA each ----
#pragma unroll
    for (int kb = 0; kb < 2; ++kb) {
      v16bf afr[4], bfr[4];
#pragma unroll
      for (int mi = 0; mi < 4; ++mi)
        afr[mi] = ld_frag72(As + kb * 32, wm * 64 + mi * 16 + l15, hi);
#pragma unroll
      for (int ni = 0; ni < 4; ++ni)
        bfr[ni] = ld_frag72(Bs + kb * 32, wn * 64 + ni * 16 + l15, hi);
#pragma unroll
      for (int mi = 0; mi < 4; ++mi)
#pragma unroll
        for (int ni = 0; ni < 4; ++ni)
          acc[mi][ni] = __builtin_amdgcn_wmma_f32_16x16x32_bf16(
              false, afr[mi], false, bfr[ni], (short)0, acc[mi][ni], false, false);
    }
    __syncthreads();
  }

  // ---- epilogue: VGPR r -> row = base + hi*8 + r, col = base + l15 ----
  unsigned short* Ch = nullptr;
  float* Cf = nullptr;
  if constexpr (OUT_BF16)
    Ch = (unsigned short*)Cp + (long long)bz * sCb;
  else
    Cf = (float*)Cp + (long long)bz * sCb;

#pragma unroll
  for (int mi = 0; mi < 4; ++mi) {
#pragma unroll
    for (int ni = 0; ni < 4; ++ni) {
      int row0 = by * 256 + wm * 64 + mi * 16 + hi * 8;
      int col = bx * 128 + wn * 64 + ni * 16 + l15;
#pragma unroll
      for (int r = 0; r < 8; ++r) {
        float v = acc[mi][ni][r] * alpha;
        if constexpr (OUT_BF16)
          Ch[(long long)(row0 + r) * ldc + col] = f2bf(v);
        else
          Cf[(long long)(row0 + r) * ldc + col] = v;
      }
    }
  }
}

// ---------------------------------------------------------------------------
// In-place row softmax over bf16 rows of width 2048. One 256-thread block per
// row; 8 values per thread; wave32 shuffle + LDS cross-wave reduction.
// ---------------------------------------------------------------------------
__global__ __launch_bounds__(256) void softmax_rows(unsigned short* __restrict__ S) {
  const int tid = threadIdx.x;
  unsigned short* row = S + (unsigned long long)blockIdx.x * 2048ull;

  float v[8];
  float mx = -3.0e38f;
#pragma unroll
  for (int i = 0; i < 8; ++i) {
    v[i] = bf2f(row[tid + i * 256]);
    mx = fmaxf(mx, v[i]);
  }
#pragma unroll
  for (int off = 16; off > 0; off >>= 1) mx = fmaxf(mx, __shfl_xor(mx, off, 32));

  __shared__ float redm[8], reds[8];
  const int wave = tid >> 5, lane = tid & 31;
  if (lane == 0) redm[wave] = mx;
  __syncthreads();
  mx = redm[0];
#pragma unroll
  for (int w = 1; w < 8; ++w) mx = fmaxf(mx, redm[w]);

  float sum = 0.0f;
#pragma unroll
  for (int i = 0; i < 8; ++i) {
    v[i] = __expf(v[i] - mx);
    sum += v[i];
  }
#pragma unroll
  for (int off = 16; off > 0; off >>= 1) sum += __shfl_xor(sum, off, 32);
  if (lane == 0) reds[wave] = sum;
  __syncthreads();
  float tot = 0.0f;
#pragma unroll
  for (int w = 0; w < 8; ++w) tot += reds[w];
  float inv = 1.0f / tot;
#pragma unroll
  for (int i = 0; i < 8; ++i) row[tid + i * 256] = f2bf(v[i] * inv);
}

// ---------------------------------------------------------------------------
// Orchestration.
//  d_in: x[4,2048,1024] f32, y[4,2048,1024] f32, Wq[1024,1024] f32,
//        Wkv[1024,2048] f32.  d_out: out[4,2048,1024] f32.
//  ws: q bf16 [8192,1024] | kv bf16 [8192,2048] | S/P bf16 [4,2048,2048]
// ---------------------------------------------------------------------------
extern "C" void kernel_launch(void* const* d_in, const int* in_sizes, int n_in,
                              void* d_out, int out_size, void* d_ws, size_t ws_size,
                              hipStream_t stream) {
  (void)in_sizes; (void)n_in; (void)out_size; (void)ws_size;
  const float* x   = (const float*)d_in[0];
  const float* y   = (const float*)d_in[1];
  const float* Wq  = (const float*)d_in[2];
  const float* Wkv = (const float*)d_in[3];
  float* out = (float*)d_out;

  char* ws = (char*)d_ws;
  unsigned short* qb  = (unsigned short*)ws;                             // 16 MiB
  unsigned short* kvb = (unsigned short*)(ws + 16777216LL);              // 32 MiB
  unsigned short* sb  = (unsigned short*)(ws + 16777216LL + 33554432LL); // 32 MiB

  const dim3 blk(256);
  const float scale = 0.03125f;  // 1024^-0.5, folded into q projection

  // q = (x @ Wq) * scale -> bf16
  gemm_wmma<1, 1, 0, 1><<<dim3(8, 32, 1), blk, 0, stream>>>(
      x, 0LL, 1024, Wq, 0LL, 1024, qb, 0LL, 1024, 1024, scale);

  // kv = y @ Wkv -> bf16 (cols 0..1023 = k, 1024..2047 = v)
  gemm_wmma<1, 1, 0, 1><<<dim3(16, 32, 1), blk, 0, stream>>>(
      y, 0LL, 1024, Wkv, 0LL, 2048, kvb, 0LL, 2048, 1024, 1.0f);

  // S = q @ k^T per batch -> bf16 (k stored [N][K] wrt this GEMM: BT=1)
  gemm_wmma<0, 0, 1, 1><<<dim3(16, 8, 4), blk, 0, stream>>>(
      qb, 2048LL * 1024LL, 1024, kvb, 2048LL * 2048LL, 2048,
      sb, 2048LL * 2048LL, 2048, 1024, 1.0f);

  // softmax rows of S in place (4*2048 rows of 2048)
  softmax_rows<<<dim3(8192), blk, 0, stream>>>(sb);

  // out = P @ v -> fp32
  gemm_wmma<0, 0, 0, 0><<<dim3(8, 8, 4), blk, 0, stream>>>(
      sb, 2048LL * 2048LL, 2048, kvb + 1024, 2048LL * 2048LL, 2048,
      out, 2048LL * 1024LL, 1024, 2048, 1.0f);
}
